// HybridMemory_23141283791269
// MI455X (gfx1250) — matrix operand order repaired and checked by reading, not compiled, not run on hardware
//
#include <hip/hip_runtime.h>
#include <hip/hip_bf16.h>

// ---------------------------------------------------------------------------
// loss = mean_i [ logsumexp_j(f_i . bank_j / T) - (f_i . bank_{pid_i} / T) ]
// Kernel 1: split-bf16 WMMA GEMM  logits = features @ bank^T * (1/T)
//           (f32 = hi + lo bf16; hi*hi + hi*lo + lo*hi ~= f32 accuracy)
//           Double-buffered XOR-swizzled LDS, one barrier per K-chunk.
// Kernel 2: per-row logsumexp + target gather -> per-row loss
// Kernel 3: mean over 512 rows -> d_out[0]
// ---------------------------------------------------------------------------

typedef __attribute__((ext_vector_type(16))) __bf16 v16bf;
typedef __attribute__((ext_vector_type(8)))  __bf16 v8bf;
typedef __attribute__((ext_vector_type(4)))  __bf16 v4bf;
typedef __attribute__((ext_vector_type(8)))  float  v8f;
typedef __attribute__((ext_vector_type(4)))  float  v4f;

#define NB 512              // batch rows (M)
#define ND 2048             // feature dim (K)
#define NL 15080            // bank rows (N)
#define NTILES_N 118        // ceil(15080/128)
#define NSTR (NTILES_N*128) // 15104 padded logits row stride
#define INV_TEMP 20.0f      // 1/0.05

#define TM 128
#define TN 128
#define TK 32
#define NKC (ND / TK)       // 64 K-chunks
#define LELEMS (TM * TK)    // 4096 bf16 per (array, buffer) = 8 KB

// XOR block swizzle: rows have 4x 16B blocks (8 bf16 each); physical block =
// logical block ^ (row>>2 & 3). Conflict-free b128/b64 LDS access per
// 16-lane phase with an unpadded 64B row stride.
__device__ __forceinline__ int swz(int row, int col) {
  const int bb = col >> 3;
  const int within = col & 7;
  return row * TK + (((bb ^ (row >> 2)) & 3) << 3) + within;
}

__launch_bounds__(256, 2)
__global__ void gemm_logits_kernel(const float* __restrict__ feat,
                                   const float* __restrict__ bank,
                                   float* __restrict__ logits) {
  // [array(Ah,Al,Bh,Bl)][buf 0/1][row][swizzled col] : 64 KB total
  __shared__ __attribute__((aligned(16))) __bf16 smem[8 * LELEMS];
  __bf16* const sAh = smem;
  __bf16* const sAl = smem + 2 * LELEMS;
  __bf16* const sBh = smem + 4 * LELEMS;
  __bf16* const sBl = smem + 6 * LELEMS;

  const int tid  = threadIdx.x;
  const int bn   = blockIdx.x;          // N tile
  const int bm   = blockIdx.y;          // M tile
  const int lane = tid & 31;
  const int wv   = tid >> 5;            // 8 waves
  const int wm   = wv >> 2;             // 0..1 -> 64-row slab
  const int wn   = wv & 3;              // 0..3 -> 32-col slab
  const int lr   = lane & 15;           // row/col within 16
  const int lh   = lane >> 4;           // lane half (K-phase select)

  // Global->reg staging: each thread carries 4x float4 for A and for B.
  const int arow = tid >> 3;            // base row (covers r, r+32, r+64, r+96)
  const int ac4  = tid & 7;             // float4 column within 32-wide chunk
  const v4f z4 = {0.f, 0.f, 0.f, 0.f};
  v4f ra[4], rb[4];

  auto load_chunk = [&](int kk) {
    const float* fp = feat + (size_t)(bm * TM) * ND + (size_t)kk * TK + ac4 * 4;
#pragma unroll
    for (int q = 0; q < 4; ++q) {
      const int r = arow + 32 * q;
      ra[q] = *(const v4f*)(fp + (size_t)r * ND);
    }
#pragma unroll
    for (int q = 0; q < 4; ++q) {
      const int gr = bn * TN + arow + 32 * q;
      rb[q] = (gr < NL)
            ? *(const v4f*)(bank + (size_t)gr * ND + (size_t)kk * TK + ac4 * 4)
            : z4;
    }
  };

  // Split f32 -> (hi, lo) bf16 and publish to LDS (8B ds_store each).
  auto cvt_store = [&](v4f x, __bf16* hp, __bf16* lp) {
    v4bf h, l;
#pragma unroll
    for (int e = 0; e < 4; ++e) {
      const float f  = x[e];
      const __bf16 hb = (__bf16)f;
      h[e] = hb;
      l[e] = (__bf16)(f - (float)hb);
    }
    *(v4bf*)hp = h;
    *(v4bf*)lp = l;
  };
  auto store_chunk = [&](int buf) {
    const int bo = buf * LELEMS;
#pragma unroll
    for (int q = 0; q < 4; ++q) {
      const int r = arow + 32 * q;
      const int o = bo + swz(r, ac4 * 4);
      cvt_store(ra[q], &sAh[o], &sAl[o]);
      cvt_store(rb[q], &sBh[o], &sBl[o]);
    }
  };

  // Fragment build: two aligned ds_load_b128 (swizzled addresses are
  // lane-constant -> hoisted out of the K-loop), no VALU.
  auto load_frag = [&](const __bf16* arr, int o0, int o1) -> v16bf {
    v8bf x0 = *(const v8bf*)(arr + o0);
    v8bf x1 = *(const v8bf*)(arr + o1);
    return __builtin_shufflevector(x0, x1, 0, 1, 2, 3, 4, 5, 6, 7,
                                           8, 9, 10, 11, 12, 13, 14, 15);
  };

  // A frag: lane holds row=lr, K = lh*8+0..7 (elems 0..7), 16+lh*8+0..7 (8..15)
  int aoff0[4], aoff1[4];
#pragma unroll
  for (int i = 0; i < 4; ++i) {
    const int r = wm * 64 + i * 16 + lr;
    aoff0[i] = swz(r, lh * 8);
    aoff1[i] = swz(r, 16 + lh * 8);
  }
  // B frag: lane holds col=lr, K = lh*8+0..7 (elems 0..7), 16+lh*8+0..7 (8..15)
  int boff0[2], boff1[2];
#pragma unroll
  for (int j = 0; j < 2; ++j) {
    const int r = wn * 32 + j * 16 + lr;
    boff0[j] = swz(r, lh * 8);
    boff1[j] = swz(r, 16 + lh * 8);
  }

  v8f acc[4][2];
  const v8f zacc = {0.f, 0.f, 0.f, 0.f, 0.f, 0.f, 0.f, 0.f};
#pragma unroll
  for (int i = 0; i < 4; ++i)
#pragma unroll
    for (int j = 0; j < 2; ++j) acc[i][j] = zacc;

  load_chunk(0);
  store_chunk(0);
  __syncthreads();

  for (int kk = 0; kk < NKC; ++kk) {
    const int cur = (kk & 1) * LELEMS;
    if (kk + 1 < NKC) load_chunk(kk + 1);     // global -> regs for next chunk
    if (kk + 2 < NKC) {                       // global_prefetch_b8 for chunk+2
#pragma unroll
      for (int q = 0; q < 4; ++q) {
        const int gr = bn * TN + arow + 32 * q;
        if (gr < NL)
          __builtin_prefetch(bank + (size_t)gr * ND + (size_t)(kk + 2) * TK + ac4 * 4, 0, 3);
      }
    }

    v16bf bh[2], bl[2];
#pragma unroll
    for (int j = 0; j < 2; ++j) {
      bh[j] = load_frag(sBh + cur, boff0[j], boff1[j]);
      bl[j] = load_frag(sBl + cur, boff0[j], boff1[j]);
    }

#pragma unroll
    for (int i = 0; i < 4; ++i) {
      v16bf ah = load_frag(sAh + cur, aoff0[i], aoff1[i]);
      v16bf al = load_frag(sAl + cur, aoff0[i], aoff1[i]);
#pragma unroll
      for (int j = 0; j < 2; ++j) {
        acc[i][j] = __builtin_amdgcn_wmma_f32_16x16x32_bf16(
            false, ah, false, bh[j], (short)0, acc[i][j], false, false);
        acc[i][j] = __builtin_amdgcn_wmma_f32_16x16x32_bf16(
            false, ah, false, bl[j], (short)0, acc[i][j], false, false);
        acc[i][j] = __builtin_amdgcn_wmma_f32_16x16x32_bf16(
            false, al, false, bh[j], (short)0, acc[i][j], false, false);
      }
    }

    // Publish next chunk into the OTHER buffer (overlaps with compute of
    // other waves); single barrier per chunk.
    if (kk + 1 < NKC) store_chunk((kk + 1) & 1);
    __syncthreads();
  }

  // --- store D: VGPR r -> row = base + lh*8 + r, col = base + lr ---
#pragma unroll
  for (int i = 0; i < 4; ++i) {
    const int row0 = bm * TM + wm * 64 + i * 16 + lh * 8;
#pragma unroll
    for (int j = 0; j < 2; ++j) {
      const int col = bn * TN + wn * 32 + j * 16 + lr;
      if (col < NL) {
#pragma unroll
        for (int r = 0; r < 8; ++r)
          logits[(size_t)(row0 + r) * NSTR + col] = acc[i][j][r] * INV_TEMP;
      }
    }
  }
}

__global__ void row_loss_kernel(const float* __restrict__ logits,
                                const int* __restrict__ gt_labels,
                                float* __restrict__ losses) {
  const int row = blockIdx.x;
  const int tid = threadIdx.x;
  const float* lr = logits + (size_t)row * NSTR;
  __shared__ float red[256];

  float m = -3.0e38f;
  for (int c = tid; c < NL; c += 256) m = fmaxf(m, lr[c]);
  red[tid] = m;
  __syncthreads();
  for (int s = 128; s > 0; s >>= 1) {
    if (tid < s) red[tid] = fmaxf(red[tid], red[tid + s]);
    __syncthreads();
  }
  m = red[0];
  __syncthreads();

  float sum = 0.f;
  for (int c = tid; c < NL; c += 256) sum += __expf(lr[c] - m);
  red[tid] = sum;
  __syncthreads();
  for (int s = 128; s > 0; s >>= 1) {
    if (tid < s) red[tid] += red[tid + s];
    __syncthreads();
  }

  if (tid == 0) {
    const int pid = gt_labels[row * 5 + 4];   // gt_labels[:, -1]
    losses[row] = m + __logf(red[0]) - lr[pid];
  }
}

__global__ void mean_kernel(const float* __restrict__ losses,
                            float* __restrict__ out) {
  const int tid = threadIdx.x;
  __shared__ float red[256];
  float s = 0.f;
  for (int i = tid; i < NB; i += 256) s += losses[i];
  red[tid] = s;
  __syncthreads();
  for (int st = 128; st > 0; st >>= 1) {
    if (tid < st) red[tid] += red[tid + st];
    __syncthreads();
  }
  if (tid == 0) out[0] = red[0] / (float)NB;
}

extern "C" void kernel_launch(void* const* d_in, const int* in_sizes, int n_in,
                              void* d_out, int out_size, void* d_ws, size_t ws_size,
                              hipStream_t stream) {
  const float* features = (const float*)d_in[0];   // (512, 2048)
  // d_in[1] features_k unused by the reference result path
  const int*   gt       = (const int*)d_in[2];     // (512, 5)
  // d_in[3] gt_labels_k unused
  const float* bank     = (const float*)d_in[4];   // (15080, 2048)

  float* logits = (float*)d_ws;                          // 512 x 15104 f32
  float* losses = logits + (size_t)NB * NSTR;            // 512 f32

  dim3 grid(NTILES_N, NB / TM);                          // 118 x 4
  gemm_logits_kernel<<<grid, 256, 0, stream>>>(features, bank, logits);
  row_loss_kernel<<<NB, 256, 0, stream>>>(logits, gt, losses);
  mean_kernel<<<1, 256, 0, stream>>>(losses, (float*)d_out);
}
